// NucleusSegmenter_78469052498366
// MI455X (gfx1250) — matrix operand order repaired and checked by reading, not compiled, not run on hardware
//
#include <hip/hip_runtime.h>
#include <hip/hip_bf16.h>
#include <stdint.h>

#define BQ    16
#define NPTS  4096
#define NROWS (BQ * NPTS)   // 65536 points total
#define KDIM  576           // fused feature width (36 * 16)
#define FW    56            // feature map H/W

typedef __attribute__((ext_vector_type(16))) __bf16 v16bf;
typedef __attribute__((ext_vector_type(8)))  __bf16 v8bf;
typedef __attribute__((ext_vector_type(8)))  float  v8f;

// tanh-approx GELU, branch-free: 0.5x(1+tanh(u)) == x * sigmoid(2u)
// lowers to v_exp_f32 + v_rcp_f32, no EXEC divergence (unlike libm tanhf)
__device__ __forceinline__ float gelu_tanh(float x) {
    float u = 0.79788456080286535588f * (x + 0.044715f * x * x * x);
    return x * __builtin_amdgcn_rcpf(1.0f + __expf(-2.0f * u));
}

// ---- WMMA tile loaders (wave32, contiguous k = 16*half + s in both A and B) ----

// A-matrix 16x32 bf16 from row-major source (global or LDS), leading dim ld.
// lanes 0-15: row=lane,   K = {0..7} in v0-3, {8..15} in v4-7  -> two 16B loads
// lanes16-31: row=lane-16,K = {16..23} in v0-3, {24..31} in v4-7
__device__ __forceinline__ v16bf load_a_tile(const __bf16* base, int ld, int m0, int k0) {
    int lane = threadIdx.x & 31;
    int half = lane >> 4;
    int row  = m0 + (lane & 15);
    const __bf16* p = base + (size_t)row * ld + k0 + half * 16;
    union { v16bf v; v8bf h[2]; } u;
    u.h[0] = *(const v8bf*)(p);        // K local 0..7 (or 16..23)
    u.h[1] = *(const v8bf*)(p + 8);    // K local 8..15 (or 24..31)
    return u.v;
}

// B-matrix 32x16 from pre-converted bf16 weight w[N][ldk] (row-major, K-padded).
// lanes 0-15 hold K k0..k0+15 (col N = n0+lane); lanes 16-31 hold K k0+16..k0+31.
// Each lane: 16 K-contiguous bf16 = two 16B loads.
__device__ __forceinline__ v16bf load_b_tile(const __bf16* w, int ldk, int n0, int k0) {
    int lane = threadIdx.x & 31;
    int n  = n0 + (lane & 15);
    int kh = k0 + ((lane >> 4) << 4);
    const __bf16* p = w + (size_t)n * ldk + kh;
    union { v16bf v; v8bf h[2]; } u;
    u.h[0] = *(const v8bf*)(p);
    u.h[1] = *(const v8bf*)(p + 8);
    return u.v;
}

// --------------------- weight preconversion f32 -> bf16 ------------------------

__global__ void __launch_bounds__(256) convert_pad_kernel(
    const float* __restrict__ src, __bf16* __restrict__ dst,
    int Ksrc, int Kdst, int total)
{
    int idx = blockIdx.x * blockDim.x + threadIdx.x;
    if (idx >= total) return;
    int n = idx / Kdst, k = idx - n * Kdst;
    dst[idx] = (k < Ksrc) ? (__bf16)src[(size_t)n * Ksrc + k] : (__bf16)0.0f;
}

// --------------------------- convolution (VALU) -------------------------------

__global__ void __launch_bounds__(256) conv3x3_gelu_kernel(
    const float* __restrict__ in, const float* __restrict__ wgt,
    const float* __restrict__ bias, float* __restrict__ out,
    int Cin, int Cout, int Hin, int Win, int Hout, int Wout, int stride, int total)
{
    int idx = blockIdx.x * blockDim.x + threadIdx.x;
    if (idx >= total) return;
    int x  = idx % Wout;
    int y  = (idx / Wout) % Hout;
    int co = (idx / (Wout * Hout)) % Cout;
    int b  = idx / (Wout * Hout * Cout);
    float acc = bias[co];
    for (int ci = 0; ci < Cin; ++ci) {
        const float* ip = in + (size_t)(b * Cin + ci) * Hin * Win;
        const float* wp = wgt + (size_t)(co * Cin + ci) * 9;
#pragma unroll
        for (int ky = 0; ky < 3; ++ky) {
            int iy = y * stride + ky - 1;
            if (iy < 0 || iy >= Hin) continue;
#pragma unroll
            for (int kx = 0; kx < 3; ++kx) {
                int ix = x * stride + kx - 1;
                if (ix < 0 || ix >= Win) continue;
                acc += ip[iy * Win + ix] * wp[ky * 3 + kx];
            }
        }
    }
    out[idx] = gelu_tanh(acc);
}

__global__ void __launch_bounds__(256) pool_kernel(const float* __restrict__ g,
                                                   float* __restrict__ gc, int total)
{
    int idx = blockIdx.x * blockDim.x + threadIdx.x;
    if (idx >= total) return;
    float s = 0.0f;
    for (int i = 0; i < 49; ++i) s += g[(size_t)idx * 49 + i];
    gc[idx] = s * (1.0f / 49.0f);
}

// ------------------- point branch: embed + MLP via WMMA ------------------------

__global__ void __launch_bounds__(256) point_mlp_kernel(
    const float* __restrict__ points,
    const __bf16* __restrict__ Wl1,   // [256][64]  bf16, K zero-padded 48->64
    const float* __restrict__ lb1,
    const __bf16* __restrict__ Wl2,   // [256][256] bf16
    const float* __restrict__ lb2,
    __bf16* __restrict__ F)
{
    __shared__ __attribute__((aligned(16))) __bf16 emb[16][64];
    __shared__ __attribute__((aligned(16))) __bf16 h1[16][256];
    const int m0  = blockIdx.x * 16;
    const int tid = threadIdx.x;

    // sin/cos positional embedding: feats[i*4+q] = {sin fx, sin fy, cos fx, cos fy}
    for (int e = tid; e < 16 * 64; e += 256) {
        int r = e >> 6, c = e & 63;
        float v = 0.0f;
        if (c < 48) {
            int row = m0 + r;
            float px = points[row * 2 + 0];
            float py = points[row * 2 + 1];
            int i = c >> 2, q = c & 3;
            float s = (float)(1 << i) * ((q & 1) ? py : px);
            v = (q < 2) ? __sinf(s) : __cosf(s);
        }
        emb[r][c] = (__bf16)v;
    }
    __syncthreads();

    const int wv   = tid >> 5;
    const int lane = tid & 31;
    const int half = lane >> 4;

    // layer 1: [16,64] x [64,256] (+bias, GELU) -> LDS
    for (int ct = wv * 2; ct < wv * 2 + 2; ++ct) {
        int n0 = ct * 16;
        v8f c = {};
#pragma unroll
        for (int ks = 0; ks < 2; ++ks) {
            v16bf a = load_a_tile(&emb[0][0], 64, 0, ks * 32);
            v16bf b = load_b_tile(Wl1, 64, n0, ks * 32);
            c = __builtin_amdgcn_wmma_f32_16x16x32_bf16(false, a, false, b,
                                                        (short)0, c, false, false);
        }
        int nn = n0 + (lane & 15);
        float bias = lb1[nn];
#pragma unroll
        for (int r = 0; r < 8; ++r)
            h1[r + 8 * half][nn] = (__bf16)gelu_tanh(c[r] + bias);
    }
    __syncthreads();

    // layer 2: [16,256] x [256,256] (+bias) -> F[:, 0:256]
    for (int ct = wv * 2; ct < wv * 2 + 2; ++ct) {
        int n0 = ct * 16;
        v8f c = {};
#pragma unroll
        for (int ks = 0; ks < 8; ++ks) {
            v16bf a = load_a_tile(&h1[0][0], 256, 0, ks * 32);
            v16bf b = load_b_tile(Wl2, 256, n0, ks * 32);
            c = __builtin_amdgcn_wmma_f32_16x16x32_bf16(false, a, false, b,
                                                        (short)0, c, false, false);
        }
        int nn = n0 + (lane & 15);
        float bias = lb2[nn];
#pragma unroll
        for (int r = 0; r < 8; ++r)
            F[(size_t)(m0 + r + 8 * half) * KDIM + nn] = (__bf16)(c[r] + bias);
    }
}

// -------------------- bilinear patch gather -> F[:, 256:512] -------------------

__global__ void __launch_bounds__(64) patch_kernel(
    const float* __restrict__ fmap, const float* __restrict__ points,
    __bf16* __restrict__ F)
{
    const int pt = blockIdx.x;               // 0 .. B*N-1
    const int b  = pt >> 12;                 // N = 4096
    const float cx = points[pt * 2 + 0] * 2.0f - 1.0f;
    const float cy = points[pt * 2 + 1] * 2.0f - 1.0f;
    const int t = threadIdx.x;               // 0..63
    const int i = t >> 3, j = t & 7;         // i: x-offset idx, j: y-offset idx
    const float step = 2.0f / (float)(FW - 1);
    float gx = cx + ((float)i - 3.5f) * step;
    float gy = cy + ((float)j - 3.5f) * step;
    float ix = (gx + 1.0f) * 0.5f * (float)(FW - 1);   // align_corners=True
    float iy = (gy + 1.0f) * 0.5f * (float)(FW - 1);
    float x0f = floorf(ix), y0f = floorf(iy);
    int x0 = (int)x0f, y0 = (int)y0f, x1 = x0 + 1, y1 = y0 + 1;
    float wx1 = ix - x0f, wx0 = 1.0f - wx1;
    float wy1 = iy - y0f, wy0 = 1.0f - wy1;

    float w00 = wx0 * wy0, w10 = wx1 * wy0, w01 = wx0 * wy1, w11 = wx1 * wy1;
    bool vx0 = (x0 >= 0) & (x0 < FW), vx1 = (x1 >= 0) & (x1 < FW);
    bool vy0 = (y0 >= 0) & (y0 < FW), vy1 = (y1 >= 0) & (y1 < FW);
    int cx0 = min(max(x0, 0), FW - 1), cx1 = min(max(x1, 0), FW - 1);
    int cy0 = min(max(y0, 0), FW - 1), cy1 = min(max(y1, 0), FW - 1);
    w00 *= (vx0 && vy0) ? 1.0f : 0.0f;
    w10 *= (vx1 && vy0) ? 1.0f : 0.0f;
    w01 *= (vx0 && vy1) ? 1.0f : 0.0f;
    w11 *= (vx1 && vy1) ? 1.0f : 0.0f;

#pragma unroll
    for (int c = 0; c < 4; ++c) {
        const float* fp = fmap + (size_t)(b * 4 + c) * FW * FW;
        float v = fp[cy0 * FW + cx0] * w00 + fp[cy0 * FW + cx1] * w10
                + fp[cy1 * FW + cx0] * w01 + fp[cy1 * FW + cx1] * w11;
        F[(size_t)pt * KDIM + 256 + c * 64 + t] = (__bf16)v;
    }
}

__global__ void __launch_bounds__(256) gc_bcast_kernel(const float* __restrict__ gc,
                                                       __bf16* __restrict__ F, int total)
{
    int idx = blockIdx.x * blockDim.x + threadIdx.x;
    if (idx >= total) return;
    int row = idx >> 6, c = idx & 63;
    int b = row >> 12;
    F[(size_t)row * KDIM + 512 + c] = (__bf16)gc[b * 64 + c];
}

__global__ void __launch_bounds__(256) out_init_kernel(float* __restrict__ out,
                                                       const float* __restrict__ hb2, int total)
{
    int idx = blockIdx.x * blockDim.x + threadIdx.x;
    if (idx < total) out[idx] = hb2[0];
}

// ----------- head: F[65536,576] @ hw1^T -> GELU -> . hw2 (fused) ---------------

__global__ void __launch_bounds__(256) head_kernel(
    const __bf16* __restrict__ F,
    const __bf16* __restrict__ Wh1,   // [512][576] bf16
    const float* __restrict__ hb1,
    const float* __restrict__ hw2, float* __restrict__ out)
{
    __shared__ __attribute__((aligned(16))) __bf16 As[16][KDIM];   // 18 KB
    const int m0   = blockIdx.x * 16;
    const int tid  = threadIdx.x;
    const int wv   = tid >> 5;
    const int lane = tid & 31;
    const int half = lane >> 4;

    // stage this block's 16 x 576 A panel into LDS (16B chunks, all 256 threads)
    for (int e = tid; e < 16 * (KDIM / 8); e += 256) {
        int r = e / (KDIM / 8), cc = e - r * (KDIM / 8);
        *(v8bf*)&As[r][cc * 8] = *(const v8bf*)&F[(size_t)(m0 + r) * KDIM + cc * 8];
    }
    __syncthreads();

    for (int q = 0; q < 4; ++q) {            // 8 waves x 4 tiles = 512 hidden cols
        int n0 = (wv * 4 + q) * 16;
        v8f c = {};
#pragma unroll
        for (int ks = 0; ks < 18; ++ks) {    // K = 576
            v16bf a = load_a_tile(&As[0][0], KDIM, 0, ks * 32);
            v16bf b = load_b_tile(Wh1, KDIM, n0, ks * 32);
            c = __builtin_amdgcn_wmma_f32_16x16x32_bf16(false, a, false, b,
                                                        (short)0, c, false, false);
        }
        int nn = n0 + (lane & 15);
        float b1 = hb1[nn];
        float w2 = hw2[nn];
#pragma unroll
        for (int r = 0; r < 8; ++r) {
            float p = gelu_tanh(c[r] + b1) * w2;
            p += __shfl_xor(p, 1);
            p += __shfl_xor(p, 2);
            p += __shfl_xor(p, 4);
            p += __shfl_xor(p, 8);           // sum over the 16 cols of this tile
            if ((lane & 15) == 0)
                atomicAdd(&out[m0 + 8 * half + r], p);
        }
    }
}

// ------------------------------- launcher --------------------------------------

extern "C" void kernel_launch(void* const* d_in, const int* in_sizes, int n_in,
                              void* d_out, int out_size, void* d_ws, size_t ws_size,
                              hipStream_t stream)
{
    const float* points = (const float*)d_in[0];
    const float* images = (const float*)d_in[1];
    const float* pw1 = (const float*)d_in[2];  const float* pb1 = (const float*)d_in[3];
    const float* pw2 = (const float*)d_in[4];  const float* pb2 = (const float*)d_in[5];
    const float* pw3 = (const float*)d_in[6];  const float* pb3 = (const float*)d_in[7];
    const float* gw1 = (const float*)d_in[8];  const float* gb1 = (const float*)d_in[9];
    const float* gw2 = (const float*)d_in[10]; const float* gb2 = (const float*)d_in[11];
    const float* gw3 = (const float*)d_in[12]; const float* gb3 = (const float*)d_in[13];
    const float* gw4 = (const float*)d_in[14]; const float* gb4 = (const float*)d_in[15];
    const float* gw5 = (const float*)d_in[16]; const float* gb5 = (const float*)d_in[17];
    const float* lw1 = (const float*)d_in[18]; const float* lb1 = (const float*)d_in[19];
    const float* lw2 = (const float*)d_in[20]; const float* lb2 = (const float*)d_in[21];
    const float* hw1 = (const float*)d_in[22]; const float* hb1 = (const float*)d_in[23];
    const float* hw2 = (const float*)d_in[24]; const float* hb2 = (const float*)d_in[25];
    float* out = (float*)d_out;

    char* ws = (char*)d_ws;
    size_t off = 0;
    auto wsalloc = [&](size_t bytes) -> void* {
        void* p = ws + off;
        off = (off + bytes + 255) & ~(size_t)255;
        return p;
    };
    float*  buf1 = (float*)wsalloc((size_t)BQ * 16 * 112 * 112 * 4);  // 12.85 MB
    float*  buf2 = (float*)wsalloc((size_t)BQ * 32 * 56 * 56 * 4);    //  6.42 MB
    float*  fmap = (float*)wsalloc((size_t)BQ * 4 * 56 * 56 * 4);     //  0.80 MB
    float*  gc   = (float*)wsalloc((size_t)BQ * 64 * 4);
    __bf16* F    = (__bf16*)wsalloc((size_t)NROWS * KDIM * 2);        // 75.5 MB
    __bf16* Wl1  = (__bf16*)wsalloc((size_t)256 * 64 * 2);            // K pad 48->64
    __bf16* Wl2  = (__bf16*)wsalloc((size_t)256 * 256 * 2);
    __bf16* Wh1  = (__bf16*)wsalloc((size_t)512 * KDIM * 2);

    // weights -> bf16 (zero-padded K), once per launch; L2-resident thereafter
    convert_pad_kernel<<<(256 * 64 + 255) / 256, 256, 0, stream>>>(lw1, Wl1, 48, 64, 256 * 64);
    convert_pad_kernel<<<(256 * 256 + 255) / 256, 256, 0, stream>>>(lw2, Wl2, 256, 256, 256 * 256);
    convert_pad_kernel<<<(512 * KDIM + 255) / 256, 256, 0, stream>>>(hw1, Wh1, KDIM, KDIM, 512 * KDIM);

    auto launch_conv = [&](const float* in, const float* w, const float* b, float* o,
                           int Cin, int Cout, int Hin, int Win, int Hout, int Wout, int s) {
        int total = BQ * Cout * Hout * Wout;
        conv3x3_gelu_kernel<<<(total + 255) / 256, 256, 0, stream>>>(
            in, w, b, o, Cin, Cout, Hin, Win, Hout, Wout, s, total);
    };

    // plane encoder -> fmap [16,4,56,56]
    launch_conv(images, pw1, pb1, buf1, 3, 16, 224, 224, 112, 112, 2);
    launch_conv(buf1,   pw2, pb2, buf2, 16, 32, 112, 112, 56, 56, 2);
    launch_conv(buf2,   pw3, pb3, fmap, 32, 4, 56, 56, 56, 56, 1);

    // global encoder -> gc [16,64]   (buf1/buf2 safely reused: stream-ordered)
    launch_conv(images, gw1, gb1, buf1, 3, 4, 224, 224, 112, 112, 2);
    launch_conv(buf1,   gw2, gb2, buf2, 4, 8, 112, 112, 56, 56, 2);
    launch_conv(buf2,   gw3, gb3, buf1, 8, 16, 56, 56, 28, 28, 2);
    launch_conv(buf1,   gw4, gb4, buf2, 16, 32, 28, 28, 14, 14, 2);
    launch_conv(buf2,   gw5, gb5, buf1, 32, 64, 14, 14, 7, 7, 2);
    pool_kernel<<<(BQ * 64 + 255) / 256, 256, 0, stream>>>(buf1, gc, BQ * 64);

    // fused feature matrix F [65536, 576] (bf16)
    point_mlp_kernel<<<NROWS / 16, 256, 0, stream>>>(points, Wl1, lb1, Wl2, lb2, F);
    patch_kernel<<<NROWS, 64, 0, stream>>>(fmap, points, F);
    {
        int total = NROWS * 64;
        gc_bcast_kernel<<<(total + 255) / 256, 256, 0, stream>>>(gc, F, total);
    }

    // head
    out_init_kernel<<<(NROWS + 255) / 256, 256, 0, stream>>>(out, hb2, NROWS);
    head_kernel<<<NROWS / 16, 256, 0, stream>>>(F, Wh1, hb1, hw2, out);
}
